// Associator_72739566125914
// MI455X (gfx1250) — matrix operand order repaired and checked by reading, not compiled, not run on hardware
//
#include <hip/hip_runtime.h>
#include <hip/hip_bf16.h>
#include <stdint.h>

typedef __attribute__((ext_vector_type(16))) _Float16 v16h;
typedef __attribute__((ext_vector_type(8)))  _Float16 v8h;
typedef __attribute__((ext_vector_type(8)))  float    v8f;
typedef _Float16 h16;

// ---------------------------------------------------------------- constants
#define C_D     256
#define C_H     4
#define C_HD    64
#define C_NTF   16384   /* fuser tokens: 1024 seqs x 16 */
#define C_NTTR  1024    /* track tokens: 4 x 256 */
#define C_NTDET 4096    /* det tokens: 4 x 1024 */

// ---------------------------------------------------------------- helpers
__device__ inline float wave_max(float v) {
#pragma unroll
  for (int m = 16; m > 0; m >>= 1) v = fmaxf(v, __shfl_xor(v, m, 32));
  return v;
}
__device__ inline float wave_sum(float v) {
#pragma unroll
  for (int m = 16; m > 0; m >>= 1) v += __shfl_xor(v, m, 32);
  return v;
}

__device__ inline v16h mk16(v8h lo, v8h hi) {
  return __builtin_shufflevector(lo, hi, 0, 1, 2, 3, 4, 5, 6, 7,
                                 8, 9, 10, 11, 12, 13, 14, 15);
}

// async global -> LDS, 16 bytes/lane (GVS mode: sgpr base + 32-bit vgpr offset)
__device__ inline void async_b128(unsigned lds_off, const void* base, int voff_bytes) {
  asm volatile("global_load_async_to_lds_b128 %0, %1, %2"
               :: "v"(lds_off), "v"(voff_bytes), "s"(base)
               : "memory");
}
__device__ inline void async_wait0() {
  asm volatile("s_wait_asynccnt 0" ::: "memory");
}

// ---------------------------------------------------------------- GEMM (WMMA)
// C[M,N] = alpha * A[M,K] x B[K,N] (+bias[m]) (+resid) (relu?) -> f32 and/or f16
// Strided: A elem (m,k) at A[m*a_rs + k*a_ks]; B elem (k,n) at B[k*b_ks + n*b_ns].
// Two-level batching: z -> (outer = z/innerB, inner = z%innerB).
struct GemmP {
  const h16* A; const h16* Bm;
  float* C; h16* Ch;
  const float* bias; const float* resid;
  long long aB0, aB1, bB0, bB1, cB0, cB1, chB0, chB1, rB0, rB1;
  int innerB;
  int a_rs, a_ks, b_ks, b_ns, c_ld, ch_ld, r_ld;
  int M, N, K;
  float alpha; int relu;
};

__global__ __launch_bounds__(128) void gemm_wmma(GemmP p) {
  // Double-buffered tiles. A m-major (k contiguous), B n-major (k contiguous);
  // 80B row pitch keeps every fragment slice 16B aligned.
  __shared__ __align__(16) h16 As[2][64][40];
  __shared__ __align__(16) h16 Bs[2][64][40];

  const int z  = blockIdx.z;
  const int bo = z / p.innerB, bi = z % p.innerB;
  const h16* A  = p.A  + bo * p.aB0 + bi * p.aB1;
  const h16* Bm = p.Bm + bo * p.bB0 + bi * p.bB1;

  const int m0 = blockIdx.y * 64, n0 = blockIdx.x * 64;
  const int tid  = threadIdx.x;
  const int wid  = tid >> 5, lane = tid & 31;
  const int g    = lane >> 4, ml = lane & 15;
  const int wm   = (wid >> 1) * 32;   // 0 or 32
  const int wn   = (wid & 1) * 32;    // 0 or 32

  const bool fullM = (m0 + 64 <= p.M);
  const bool fullN = (n0 + 64 <= p.N);

  // ---- staging (uniform path selection per block)
  auto stageA = [&](int kk, int bufi) {
    const bool fullK = (kk + 32 <= p.K);
    if (p.a_ks == 1 && fullM && fullK) {
      // k-contiguous: async DMA straight into LDS, 2 x 16B per lane
      int r = tid >> 1, c = (tid & 1) * 16;
      int va = ((m0 + r) * p.a_rs + (kk + c)) * 2;
      async_b128((unsigned)(uintptr_t)&As[bufi][r][c],     (const void*)A, va);
      async_b128((unsigned)(uintptr_t)&As[bufi][r][c + 8], (const void*)A, va + 16);
    } else if (p.a_rs == 1 && fullM && fullK) {
      // m-contiguous (transposed A): vector load along M, transpose on LDS store
#pragma unroll
      for (int i = 0; i < 2; ++i) {
        int v = tid + i * 128;
        int vm = (v & 7) * 8, vk = v >> 3;
        v8h d = *(const v8h*)(A + (m0 + vm) + (long long)(kk + vk) * p.a_ks);
#pragma unroll
        for (int jj = 0; jj < 8; ++jj) As[bufi][vm + jj][vk] = d[jj];
      }
    } else {
      // generic edge path: branch-free clamped loads
#pragma unroll
      for (int i = 0; i < 16; ++i) {
        int e = tid + i * 128;
        int r = e >> 5, c = e & 31;
        int mg = m0 + r, kg = kk + c;
        int mc = mg < p.M ? mg : p.M - 1;
        int kc = kg < p.K ? kg : p.K - 1;
        h16 v = A[(long long)mc * p.a_rs + (long long)kc * p.a_ks];
        As[bufi][r][c] = (mg < p.M && kg < p.K) ? v : (h16)0.0f;
      }
    }
  };
  auto stageB = [&](int kk, int bufi) {
    const bool fullK = (kk + 32 <= p.K);
    if (p.b_ks == 1 && fullN && fullK) {
      int nn = tid >> 1, c = (tid & 1) * 16;
      int vb = ((n0 + nn) * p.b_ns + (kk + c)) * 2;
      async_b128((unsigned)(uintptr_t)&Bs[bufi][nn][c],     (const void*)Bm, vb);
      async_b128((unsigned)(uintptr_t)&Bs[bufi][nn][c + 8], (const void*)Bm, vb + 16);
    } else if (p.b_ns == 1 && fullN && fullK) {
      // n-contiguous: vector load along N, transpose on LDS store
#pragma unroll
      for (int i = 0; i < 2; ++i) {
        int v = tid + i * 128;
        int vn = (v & 7) * 8, vk = v >> 3;
        v8h d = *(const v8h*)(Bm + (long long)(kk + vk) * p.b_ks + (n0 + vn));
#pragma unroll
        for (int jj = 0; jj < 8; ++jj) Bs[bufi][vn + jj][vk] = d[jj];
      }
    } else {
#pragma unroll
      for (int i = 0; i < 16; ++i) {
        int e = tid + i * 128;
        int kk2 = e >> 6, nn = e & 63;
        int kg = kk + kk2, ng = n0 + nn;
        int kc = kg < p.K ? kg : p.K - 1;
        int nc = ng < p.N ? ng : p.N - 1;
        h16 v = Bm[(long long)kc * p.b_ks + (long long)nc * p.b_ns];
        Bs[bufi][nn][kk2] = (kg < p.K && ng < p.N) ? v : (h16)0.0f;
      }
    }
  };

  v8f acc[2][2] = {};

  // ---- software pipeline: stage(t+1) overlaps WMMA(t)
  stageA(0, 0);
  stageB(0, 0);
  async_wait0();
  __syncthreads();

  int pb = 0;
  for (int k0 = 0; k0 < p.K; k0 += 32, pb ^= 1) {
    if (k0 + 32 < p.K) {
      stageA(k0 + 32, pb ^ 1);
      stageB(k0 + 32, pb ^ 1);
    }

    // fragments (ISA 7.12.2): A rows m-major; B rows n-major
    v16h af[2], bf[2];
#pragma unroll
    for (int s = 0; s < 2; ++s) {
      const h16* ar = &As[pb][wm + s * 16 + ml][0];
      // A half j<8 -> k = g*8+j ; j>=8 -> k = 16+g*8+(j-8)
      af[s] = mk16(*(const v8h*)(ar + g * 8), *(const v8h*)(ar + 16 + g * 8));
      const h16* br = &Bs[pb][wn + s * 16 + ml][0];
      // B half j<8 -> k = g*8+j ; j>=8 -> k = 16+g*8+(j-8)
      bf[s] = mk16(*(const v8h*)(br + g * 8), *(const v8h*)(br + 16 + g * 8));
    }
#pragma unroll
    for (int sm = 0; sm < 2; ++sm)
#pragma unroll
      for (int sn = 0; sn < 2; ++sn)
        acc[sm][sn] = __builtin_amdgcn_wmma_f32_16x16x32_f16(
            false, af[sm], false, bf[sn], (short)0, acc[sm][sn], false, false);

    async_wait0();      // drain DMA into the other buffer
    __syncthreads();    // all waves done reading buf[pb]; writes visible
  }

  // ---------------- epilogue
  float* C  = p.C  ? p.C  + bo * p.cB0  + bi * p.cB1  : (float*)0;
  h16*   Ch = p.Ch ? p.Ch + bo * p.chB0 + bi * p.chB1 : (h16*)0;
  const float* R = p.resid ? p.resid + bo * p.rB0 + bi * p.rB1 : (const float*)0;

#pragma unroll
  for (int sn = 0; sn < 2; ++sn) {
    int n = n0 + wn + sn * 16 + ml;
    if (n < p.N) {
#pragma unroll
      for (int sm = 0; sm < 2; ++sm) {
#pragma unroll
        for (int r = 0; r < 8; ++r) {
          int m = m0 + wm + sm * 16 + r + g * 8;
          if (m < p.M) {
            float val = acc[sm][sn][r] * p.alpha;
            if (p.bias) val += p.bias[m];
            if (R)      val += R[(long long)m * p.r_ld + n];
            if (p.relu) val = fmaxf(val, 0.0f);
            if (C)  C[(long long)m * p.c_ld + n] = val;
            if (Ch) Ch[(long long)m * p.ch_ld + n] = (h16)val;
          }
        }
      }
    }
  }
}

// ---------------------------------------------------------------- elementwise
__global__ void cast16_k(const float* s, h16* d, long long n) {
  for (long long i = blockIdx.x * (long long)blockDim.x + threadIdx.x; i < n;
       i += (long long)gridDim.x * blockDim.x)
    d[i] = (h16)s[i];
}

__global__ void addcast_k(float* x32, h16* x16, const float* delta, long long n) {
  for (long long i = blockIdx.x * (long long)blockDim.x + threadIdx.x; i < n;
       i += (long long)gridDim.x * blockDim.x) {
    float v = x32[i] + delta[i];
    x32[i] = v; x16[i] = (h16)v;
  }
}

// extract feature rows 1..256 of [batch, 257, tpb] into [256, ntok] f16
__global__ void pack_feats_k(const float* src, h16* dst, int tpb, long long bStride, long long ntok) {
  long long total = 256LL * ntok;
  for (long long i = blockIdx.x * (long long)blockDim.x + threadIdx.x; i < total;
       i += (long long)gridDim.x * blockDim.x) {
    long long c = i / ntok, t = i % ntok;
    long long bb = t / tpb, ti = t % tpb;
    dst[i] = (h16)src[bb * bStride + (1 + c) * tpb + ti];
  }
}

// x += positional encoding computed from position row (detached in reference)
__global__ void posenc_add_k(float* x32, h16* x16, const float* pos, long long posBStride,
                             int tpb, long long ntok) {
  long long total = 256LL * ntok;
  for (long long i = blockIdx.x * (long long)blockDim.x + threadIdx.x; i < total;
       i += (long long)gridDim.x * blockDim.x) {
    int c = (int)(i / ntok); long long t = i % ntok;
    long long bb = t / tpb, ti = t % tpb;
    float p = pos[bb * posBStride + ti];
    int half = c >> 1;
    float div = __expf((float)(2 * half) * (-9.210340371976184f / 256.0f)); // -ln(1e4)/D
    float ang = p * div;
    float pe = (c & 1) ? __cosf(ang) : __sinf(ang);
    float v = x32[i] + pe;
    x32[i] = v; x16[i] = (h16)v;
  }
}

// row softmax: S [batch][rows][cols] f32 -> P f16, one wave per row
__global__ void softmax_rows_k(const float* S, h16* P, int rows, int cols,
                               long long sB, long long pB) {
  const float* Sb = S + (long long)blockIdx.z * sB;
  h16* Pb = P + (long long)blockIdx.z * pB;
  int wid = threadIdx.x >> 5, lane = threadIdx.x & 31;
  int row = blockIdx.x * (blockDim.x >> 5) + wid;
  if (row >= rows) return;
  const float* r = Sb + (long long)row * cols;
  float mx = -1e30f;
  for (int c = lane; c < cols; c += 32) mx = fmaxf(mx, r[c]);
  mx = wave_max(mx);
  float sm = 0.0f;
  for (int c = lane; c < cols; c += 32) sm += __expf(r[c] - mx);
  sm = wave_sum(sm);
  float inv = 1.0f / sm;
  for (int c = lane; c < cols; c += 32)
    Pb[(long long)row * cols + c] = (h16)(__expf(r[c] - mx) * inv);
}

// mean over T=16 : x [256, 1024*16] -> o [256, 1024]
__global__ void meanpool_k(const float* x, float* o32, h16* o16) {
  long long total = 256LL * 1024;
  for (long long i = blockIdx.x * (long long)blockDim.x + threadIdx.x; i < total;
       i += (long long)gridDim.x * blockDim.x) {
    long long c = i >> 10, s = i & 1023;
    const float* p = x + c * C_NTF + s * 16;
    float acc = 0.0f;
#pragma unroll
    for (int t = 0; t < 16; ++t) acc += p[t];
    acc *= (1.0f / 16.0f);
    o32[i] = acc; o16[i] = (h16)acc;
  }
}

// Z0 [4, 257, 1025] from scores [4,256,1024] and bin alpha
__global__ void build_z_k(const float* sc, const float* alpha_p, float* Z) {
  float a = *alpha_p;
  long long total = 4LL * 257 * 1025;
  for (long long i = blockIdx.x * (long long)blockDim.x + threadIdx.x; i < total;
       i += (long long)gridDim.x * blockDim.x) {
    long long b = i / (257LL * 1025);
    int r = (int)((i / 1025) % 257);
    int c = (int)(i % 1025);
    Z[i] = (r < 256 && c < 1024) ? sc[(b * 256 + r) * 1024 + c] : a;
  }
}

// persistent Sinkhorn: one block per batch; u,v in LDS; 100 iterations
__global__ __launch_bounds__(1024) void sinkhorn_k(const float* Z0, float* out, int iters) {
  const float* Z = Z0 + (long long)blockIdx.x * 257 * 1025;
  float* O = out + (long long)blockIdx.x * 257 * 1025;
  __shared__ float u[257];
  __shared__ float vv[1025];
  const float norm = -logf(1280.0f);                 // -log(m+n)
  const float lmu_bin = logf(1024.0f) + norm;        // log(ns)+norm
  const float lnu_bin = logf(256.0f) + norm;         // log(ms)+norm
  for (int i = threadIdx.x; i < 1025; i += blockDim.x) { vv[i] = 0.0f; if (i < 257) u[i] = 0.0f; }
  __syncthreads();
  int wid = threadIdx.x >> 5, lane = threadIdx.x & 31, nw = blockDim.x >> 5;
  for (int it = 0; it < iters; ++it) {
    for (int r = wid; r < 257; r += nw) {
      const float* zr = Z + (long long)r * 1025;
      float mx = -1e30f;
      for (int c = lane; c < 1025; c += 32) mx = fmaxf(mx, zr[c] + vv[c]);
      mx = wave_max(mx);
      float s = 0.0f;
      for (int c = lane; c < 1025; c += 32) s += __expf(zr[c] + vv[c] - mx);
      s = wave_sum(s);
      if (lane == 0) u[r] = ((r < 256) ? norm : lmu_bin) - (mx + __logf(s));
    }
    __syncthreads();
    for (int c = wid; c < 1025; c += nw) {
      float mx = -1e30f;
      for (int r = lane; r < 257; r += 32) mx = fmaxf(mx, Z[(long long)r * 1025 + c] + u[r]);
      mx = wave_max(mx);
      float s = 0.0f;
      for (int r = lane; r < 257; r += 32) s += __expf(Z[(long long)r * 1025 + c] + u[r] - mx);
      s = wave_sum(s);
      if (lane == 0) vv[c] = ((c < 1024) ? norm : lnu_bin) - (mx + __logf(s));
    }
    __syncthreads();
  }
  for (long long i = threadIdx.x; i < 257LL * 1025; i += blockDim.x) {
    int r = (int)(i / 1025), c = (int)(i % 1025);
    O[i] = Z[i] + u[r] + vv[c] - norm;
  }
}

// ---------------------------------------------------------------- host
extern "C" void kernel_launch(void* const* d_in, const int* in_sizes, int n_in,
                              void* d_out, int out_size, void* d_ws, size_t ws_size,
                              hipStream_t stream) {
  (void)in_sizes; (void)n_in; (void)out_size; (void)ws_size;
  const float* detections = (const float*)d_in[0];
  const float* tracks     = (const float*)d_in[1];
  const float* enc_w1 = (const float*)d_in[2];  const float* enc_b1 = (const float*)d_in[3];
  const float* enc_w2 = (const float*)d_in[4];  const float* enc_b2 = (const float*)d_in[5];
  const float* fus_pw = (const float*)d_in[6];  const float* fus_pb = (const float*)d_in[7];
  const float* fus_mw = (const float*)d_in[8];  const float* fus_mb = (const float*)d_in[9];
  const float* fus_m1w = (const float*)d_in[10]; const float* fus_m1b = (const float*)d_in[11];
  const float* fus_m2w = (const float*)d_in[12]; const float* fus_m2b = (const float*)d_in[13];
  const float* gnn_pw = (const float*)d_in[14]; const float* gnn_pb = (const float*)d_in[15];
  const float* gnn_mw = (const float*)d_in[16]; const float* gnn_mb = (const float*)d_in[17];
  const float* gnn_m1w = (const float*)d_in[18]; const float* gnn_m1b = (const float*)d_in[19];
  const float* gnn_m2w = (const float*)d_in[20]; const float* gnn_m2b = (const float*)d_in[21];
  const float* final_w = (const float*)d_in[22]; const float* final_b = (const float*)d_in[23];
  const float* bin_score = (const float*)d_in[24];
  float* out = (float*)d_out;

  // bump allocator over scratch
  char* base = (char*)d_ws; size_t off = 0;
  auto allocB = [&](size_t bytes) -> void* {
    void* p = base + off; off += (bytes + 255) & ~(size_t)255; return p;
  };
  auto allocH = [&](size_t n) -> h16*   { return (h16*)allocB(n * sizeof(h16)); };
  auto allocF = [&](size_t n) -> float* { return (float*)allocB(n * sizeof(float)); };

  auto cast = [&](const float* s, h16* d, long long n) {
    cast16_k<<<dim3(1024), dim3(256), 0, stream>>>(s, d, n);
  };

  // f16 weight copies
  h16* Henc1 = allocH(65536);              cast(enc_w1, Henc1, 65536);
  h16* Henc2 = allocH(65536);              cast(enc_w2, Henc2, 65536);
  h16* Hfpw  = allocH(4LL * 3 * 65536);    cast(fus_pw, Hfpw, 4LL * 3 * 65536);
  h16* Hfmw  = allocH(4LL * 65536);        cast(fus_mw, Hfmw, 4LL * 65536);
  h16* Hfm1  = allocH(4LL * 262144);       cast(fus_m1w, Hfm1, 4LL * 262144);
  h16* Hfm2  = allocH(4LL * 131072);       cast(fus_m2w, Hfm2, 4LL * 131072);
  h16* Hgpw  = allocH(12LL * 3 * 65536);   cast(gnn_pw, Hgpw, 12LL * 3 * 65536);
  h16* Hgmw  = allocH(12LL * 65536);       cast(gnn_mw, Hgmw, 12LL * 65536);
  h16* Hgm1  = allocH(12LL * 262144);      cast(gnn_m1w, Hgm1, 12LL * 262144);
  h16* Hgm2  = allocH(12LL * 131072);      cast(gnn_m2w, Hgm2, 12LL * 131072);
  h16* Hfin  = allocH(65536);              cast(final_w, Hfin, 65536);

  // activations
  float* xf32  = allocF(256LL * C_NTF);  h16* xf16 = allocH(256LL * C_NTF);
  h16* q16 = allocH(256LL * C_NTF);
  h16* k16 = allocH(256LL * C_NTF);
  h16* v16 = allocH(256LL * C_NTF);
  float* sc32 = allocF(16LL * 1024 * 1024);
  h16*   p16  = allocH(16LL * 1024 * 1024);
  h16* msg16 = allocH(256LL * C_NTF);
  float* c132 = allocF(512LL * C_NTF);
  h16*   h116 = allocH(512LL * C_NTF);
  float* trf32  = allocF(256LL * C_NTTR);  h16* trf16  = allocH(256LL * C_NTTR);
  float* detf32 = allocF(256LL * C_NTDET); h16* detf16 = allocH(256LL * C_NTDET);
  float* dtr  = allocF(256LL * C_NTTR);
  float* ddet = allocF(256LL * C_NTDET);
  h16* m0h = allocH(256LL * C_NTTR);
  h16* m1h = allocH(256LL * C_NTDET);
  float* Z0 = allocF(4LL * 257 * 1025);

  auto run_gemm = [&](GemmP p, int batches) {
    if (p.innerB < 1) p.innerB = 1;
    dim3 g((unsigned)((p.N + 63) / 64), (unsigned)((p.M + 63) / 64), (unsigned)batches);
    gemm_wmma<<<g, dim3(128), 0, stream>>>(p);
  };
  auto mk = []() { GemmP p{}; p.innerB = 1; p.alpha = 1.0f; return p; };

  // ---------------- encoder: tracks -> fuser x ----------------
  pack_feats_k<<<dim3(1024), dim3(256), 0, stream>>>(tracks, v16, 16, 257LL * 16, (long long)C_NTF);
  { GemmP p = mk();
    p.A = Henc1; p.a_rs = 256; p.a_ks = 1;
    p.Bm = v16; p.b_ks = C_NTF; p.b_ns = 1;
    p.bias = enc_b1; p.relu = 1;
    p.Ch = q16; p.ch_ld = C_NTF;
    p.M = 256; p.N = C_NTF; p.K = 256; run_gemm(p, 1); }
  { GemmP p = mk();
    p.A = Henc2; p.a_rs = 256; p.a_ks = 1;
    p.Bm = q16; p.b_ks = C_NTF; p.b_ns = 1;
    p.bias = enc_b2;
    p.C = xf32; p.c_ld = C_NTF; p.Ch = xf16; p.ch_ld = C_NTF;
    p.M = 256; p.N = C_NTF; p.K = 256; run_gemm(p, 1); }
  // x = trk + trk_pe
  posenc_add_k<<<dim3(1024), dim3(256), 0, stream>>>(xf32, xf16, tracks, 257LL * 16, 16, (long long)C_NTF);

  // ---------------- fuser: 4 self-attention layers over T=16 ----------------
  for (int L = 0; L < 4; ++L) {
    const h16* PW = Hfpw + (size_t)L * 3 * 65536;
    const float* PB = fus_pb + (size_t)L * 3 * 256;
    // q, k, v projections (x = src)
    { GemmP p = mk();
      p.A = PW; p.a_rs = 256; p.a_ks = 1;
      p.Bm = xf16; p.b_ks = C_NTF; p.b_ns = 1;
      p.bias = PB; p.Ch = q16; p.ch_ld = C_NTF;
      p.M = 256; p.N = C_NTF; p.K = 256; run_gemm(p, 1);
      p.A = PW + 65536; p.bias = PB + 256; p.Ch = k16; run_gemm(p, 1);
      p.A = PW + 2 * 65536; p.bias = PB + 512; p.Ch = v16; run_gemm(p, 1); }
    // scores[(s,h)][16][16] = q^T k / 8  (channel = hd*H + h)
    { GemmP s = mk(); s.innerB = 4; s.alpha = 0.125f;
      s.A = q16; s.a_rs = 1; s.a_ks = 4 * C_NTF; s.aB0 = 16; s.aB1 = C_NTF;
      s.Bm = k16; s.b_ks = 4 * C_NTF; s.b_ns = 1; s.bB0 = 16; s.bB1 = C_NTF;
      s.C = sc32; s.c_ld = 16; s.cB0 = 4 * 256; s.cB1 = 256;
      s.M = 16; s.N = 16; s.K = 64; run_gemm(s, 1024 * 4); }
    softmax_rows_k<<<dim3(2, 1, 4096), dim3(256), 0, stream>>>(sc32, p16, 16, 16, 256LL, 256LL);
    // msg (reuse q16): [64,16] = v x P^T per (seq, head)
    { GemmP m = mk(); m.innerB = 4;
      m.A = v16; m.a_rs = 4 * C_NTF; m.a_ks = 1; m.aB0 = 16; m.aB1 = C_NTF;
      m.Bm = p16; m.b_ks = 1; m.b_ns = 16; m.bB0 = 4 * 256; m.bB1 = 256;
      m.Ch = q16; m.ch_ld = 4 * C_NTF; m.chB0 = 16; m.chB1 = C_NTF;
      m.M = 64; m.N = 16; m.K = 16; run_gemm(m, 4096); }
    // merge
    { GemmP g = mk();
      g.A = Hfmw + (size_t)L * 65536; g.a_rs = 256; g.a_ks = 1;
      g.Bm = q16; g.b_ks = C_NTF; g.b_ns = 1;
      g.bias = fus_mb + (size_t)L * 256;
      g.Ch = msg16; g.ch_ld = C_NTF;
      g.M = 256; g.N = C_NTF; g.K = 256; run_gemm(g, 1); }
    // MLP on concat(x, msg): split W1 = [W1a | W1b]
    { GemmP a = mk();
      a.A = Hfm1 + (size_t)L * 262144; a.a_rs = 512; a.a_ks = 1;
      a.Bm = xf16; a.b_ks = C_NTF; a.b_ns = 1;
      a.C = c132; a.c_ld = C_NTF;
      a.M = 512; a.N = C_NTF; a.K = 256; run_gemm(a, 1); }
    { GemmP b = mk();
      b.A = Hfm1 + (size_t)L * 262144 + 256; b.a_rs = 512; b.a_ks = 1;
      b.Bm = msg16; b.b_ks = C_NTF; b.b_ns = 1;
      b.resid = c132; b.r_ld = C_NTF;
      b.bias = fus_m1b + (size_t)L * 512; b.relu = 1;
      b.Ch = h116; b.ch_ld = C_NTF;
      b.M = 512; b.N = C_NTF; b.K = 256; run_gemm(b, 1); }
    // MLP2 + residual into x
    { GemmP c = mk();
      c.A = Hfm2 + (size_t)L * 131072; c.a_rs = 512; c.a_ks = 1;
      c.Bm = h116; c.b_ks = C_NTF; c.b_ns = 1;
      c.bias = fus_m2b + (size_t)L * 256;
      c.resid = xf32; c.r_ld = C_NTF;
      c.C = xf32; c.c_ld = C_NTF; c.Ch = xf16; c.ch_ld = C_NTF;
      c.M = 256; c.N = C_NTF; c.K = 512; run_gemm(c, 1); }
  }

  // mean pool over T -> tr [256, 1024]
  meanpool_k<<<dim3(512), dim3(256), 0, stream>>>(xf32, trf32, trf16);

  // ---------------- encoder: detections ----------------
  pack_feats_k<<<dim3(1024), dim3(256), 0, stream>>>(detections, v16, 1024, 257LL * 1024, (long long)C_NTDET);
  { GemmP p = mk();
    p.A = Henc1; p.a_rs = 256; p.a_ks = 1;
    p.Bm = v16; p.b_ks = C_NTDET; p.b_ns = 1;
    p.bias = enc_b1; p.relu = 1;
    p.Ch = q16; p.ch_ld = C_NTDET;
    p.M = 256; p.N = C_NTDET; p.K = 256; run_gemm(p, 1); }
  { GemmP p = mk();
    p.A = Henc2; p.a_rs = 256; p.a_ks = 1;
    p.Bm = q16; p.b_ks = C_NTDET; p.b_ns = 1;
    p.bias = enc_b2;
    p.C = detf32; p.c_ld = C_NTDET; p.Ch = detf16; p.ch_ld = C_NTDET;
    p.M = 256; p.N = C_NTDET; p.K = 256; run_gemm(p, 1); }
  posenc_add_k<<<dim3(1024), dim3(256), 0, stream>>>(detf32, detf16, detections, 257LL * 1024, 1024, (long long)C_NTDET);

  // ---------------- GNN: 12 alternating self/cross layers ----------------
  auto prop = [&](const h16* xf, int ntx, int tpx,
                  const h16* srcf, int nts, int tps,
                  const h16* PW, const float* PB, const h16* MW, const float* MB,
                  const h16* M1W, const float* M1B, const h16* M2W, const float* M2B,
                  float* delta) {
    { GemmP p = mk();
      p.A = PW; p.a_rs = 256; p.a_ks = 1;
      p.Bm = xf; p.b_ks = ntx; p.b_ns = 1;
      p.bias = PB; p.Ch = q16; p.ch_ld = ntx;
      p.M = 256; p.N = ntx; p.K = 256; run_gemm(p, 1);
      p.A = PW + 65536; p.bias = PB + 256;
      p.Bm = srcf; p.b_ks = nts; p.Ch = k16; p.ch_ld = nts; p.N = nts; run_gemm(p, 1);
      p.A = PW + 2 * 65536; p.bias = PB + 512; p.Ch = v16; run_gemm(p, 1); }
    { GemmP s = mk(); s.innerB = 4; s.alpha = 0.125f;
      s.A = q16; s.a_rs = 1; s.a_ks = 4 * ntx; s.aB0 = tpx; s.aB1 = ntx;
      s.Bm = k16; s.b_ks = 4 * nts; s.b_ns = 1; s.bB0 = tps; s.bB1 = nts;
      s.C = sc32; s.c_ld = tps; s.cB0 = 4LL * tpx * tps; s.cB1 = (long long)tpx * tps;
      s.M = tpx; s.N = tps; s.K = 64; run_gemm(s, 16); }
    softmax_rows_k<<<dim3((tpx + 7) / 8, 1, 16), dim3(256), 0, stream>>>(
        sc32, p16, tpx, tps, (long long)tpx * tps, (long long)tpx * tps);
    { GemmP m = mk(); m.innerB = 4;
      m.A = v16; m.a_rs = 4 * nts; m.a_ks = 1; m.aB0 = tps; m.aB1 = nts;
      m.Bm = p16; m.b_ks = 1; m.b_ns = tps; m.bB0 = 4LL * tpx * tps; m.bB1 = (long long)tpx * tps;
      m.Ch = q16; m.ch_ld = 4 * ntx; m.chB0 = tpx; m.chB1 = ntx;
      m.M = 64; m.N = tpx; m.K = tps; run_gemm(m, 16); }
    { GemmP g = mk();
      g.A = MW; g.a_rs = 256; g.a_ks = 1;
      g.Bm = q16; g.b_ks = ntx; g.b_ns = 1;
      g.bias = MB; g.Ch = msg16; g.ch_ld = ntx;
      g.M = 256; g.N = ntx; g.K = 256; run_gemm(g, 1); }
    { GemmP a = mk();
      a.A = M1W; a.a_rs = 512; a.a_ks = 1;
      a.Bm = xf; a.b_ks = ntx; a.b_ns = 1;
      a.C = c132; a.c_ld = ntx;
      a.M = 512; a.N = ntx; a.K = 256; run_gemm(a, 1); }
    { GemmP b = mk();
      b.A = M1W + 256; b.a_rs = 512; b.a_ks = 1;
      b.Bm = msg16; b.b_ks = ntx; b.b_ns = 1;
      b.resid = c132; b.r_ld = ntx; b.bias = M1B; b.relu = 1;
      b.Ch = h116; b.ch_ld = ntx;
      b.M = 512; b.N = ntx; b.K = 256; run_gemm(b, 1); }
    { GemmP c = mk();
      c.A = M2W; c.a_rs = 512; c.a_ks = 1;
      c.Bm = h116; c.b_ks = ntx; c.b_ns = 1;
      c.bias = M2B;
      c.C = delta; c.c_ld = ntx;
      c.M = 256; c.N = ntx; c.K = 512; run_gemm(c, 1); }
  };

  for (int i = 0; i < 12; ++i) {
    const h16* PW = Hgpw + (size_t)i * 3 * 65536;
    const float* PB = gnn_pb + (size_t)i * 3 * 256;
    const h16* MW = Hgmw + (size_t)i * 65536;  const float* MB = gnn_mb + (size_t)i * 256;
    const h16* M1W = Hgm1 + (size_t)i * 262144; const float* M1B = gnn_m1b + (size_t)i * 512;
    const h16* M2W = Hgm2 + (size_t)i * 131072; const float* M2B = gnn_m2b + (size_t)i * 256;
    const h16 *s0, *s1; int n0, t0, n1, t1;
    if (i % 2 == 1) { // cross
      s0 = detf16; n0 = C_NTDET; t0 = 1024; s1 = trf16; n1 = C_NTTR; t1 = 256;
    } else {          // self
      s0 = trf16; n0 = C_NTTR; t0 = 256; s1 = detf16; n1 = C_NTDET; t1 = 1024;
    }
    prop(trf16, C_NTTR, 256, s0, n0, t0, PW, PB, MW, MB, M1W, M1B, M2W, M2B, dtr);
    prop(detf16, C_NTDET, 1024, s1, n1, t1, PW, PB, MW, MB, M1W, M1B, M2W, M2B, ddet);
    addcast_k<<<dim3(512), dim3(256), 0, stream>>>(trf32, trf16, dtr, 256LL * C_NTTR);
    addcast_k<<<dim3(512), dim3(256), 0, stream>>>(detf32, detf16, ddet, 256LL * C_NTDET);
  }

  // ---------------- final projection + score matrix ----------------
  { GemmP p = mk();
    p.A = Hfin; p.a_rs = 256; p.a_ks = 1;
    p.Bm = trf16; p.b_ks = C_NTTR; p.b_ns = 1;
    p.bias = final_b; p.Ch = m0h; p.ch_ld = C_NTTR;
    p.M = 256; p.N = C_NTTR; p.K = 256; run_gemm(p, 1);
    p.Bm = detf16; p.b_ks = C_NTDET; p.Ch = m1h; p.ch_ld = C_NTDET; p.N = C_NTDET;
    run_gemm(p, 1); }
  { GemmP s = mk(); s.alpha = 0.0625f;   // 1/sqrt(256)
    s.A = m0h; s.a_rs = 1; s.a_ks = C_NTTR; s.aB0 = 256;
    s.Bm = m1h; s.b_ks = C_NTDET; s.b_ns = 1; s.bB0 = 1024;
    s.C = sc32; s.c_ld = 1024; s.cB0 = 256LL * 1024;
    s.M = 256; s.N = 1024; s.K = 256; run_gemm(s, 4); }

  // ---------------- Sinkhorn ----------------
  build_z_k<<<dim3(1024), dim3(256), 0, stream>>>(sc32, bin_score, Z0);
  sinkhorn_k<<<dim3(4), dim3(1024), 0, stream>>>(Z0, out, 100);
}
